// CausalSelfAttention_2877628089072
// MI455X (gfx1250) — matrix operand order repaired and checked by reading, not compiled, not run on hardware
//
#include <hip/hip_runtime.h>
#include <math.h>

#define B_ 4
#define T_ 2048
#define C_ 1024
#define H_ 16
#define D_ 64

typedef __attribute__((ext_vector_type(16))) __bf16 v16bf;
typedef __attribute__((ext_vector_type(8)))  __bf16 v8bf;
typedef __attribute__((ext_vector_type(8)))  float  v8f;

// ---------------------------------------------------------------------------
// A-fragment loader for V_WMMA_*_16X16X32_BF16 (16x32, row-major source).
// Lane L holds row (L%16); element i holds K = kb+i (i<8) / 16+kb+(i-8),
// kb = (L/16)*8  -> two contiguous 16-byte chunks per lane.
// ---------------------------------------------------------------------------
static __device__ inline v16bf load_a_frag(const __bf16* __restrict__ base,
                                           int ld, int row0, int k0, int lane) {
  const __bf16* p = base + (size_t)(row0 + (lane & 15)) * ld + (k0 + ((lane >> 4) << 3));
  v8bf lo = *(const v8bf*)p;
  v8bf hi = *(const v8bf*)(p + 16);
  return __builtin_shufflevector(lo, hi, 0,1,2,3,4,5,6,7,8,9,10,11,12,13,14,15);
}

// ---------------------------------------------------------------------------
// fp32 -> bf16 conversion
// ---------------------------------------------------------------------------
__global__ void cvt_bf16(const float* __restrict__ in, __bf16* __restrict__ out, int n) {
  int i = blockIdx.x * blockDim.x + threadIdx.x;
  if (i < n) out[i] = (__bf16)in[i];
}

// ---------------------------------------------------------------------------
// Projection GEMM: Y = A(MxC) @ W(CxC) + bias, one 32x64 tile per wave
// (two 16-row A-fragments reuse each B-fragment -> 8 WMMAs / 4 b128 + 64 16-bit loads).
// N-tile (64) == head dim D, so the N-tile index == head index.
// mode 0: Q  -> bf16 (B,H,T,D)
// mode 1: K  -> bf16 (B,H,D,T)   (transposed for QK^T B-fragments)
// mode 2: V  -> bf16 (B,H,T,D)
// mode 3: out-> fp32 (B,T,C)
// ---------------------------------------------------------------------------
__global__ void __launch_bounds__(256)
gemm_bf16_wmma(const __bf16* __restrict__ A, const __bf16* __restrict__ W,
               const float* __restrict__ bias,
               __bf16* __restrict__ dst_bf, float* __restrict__ dst_f, int mode) {
  const int lane = threadIdx.x & 31;
  const int wid  = (blockIdx.x * blockDim.x + threadIdx.x) >> 5;
  const int NT   = C_ / 64;            // 16 n-tiles
  const int m0   = (wid / NT) * 32;
  const int n0   = (wid % NT) * 64;

  v8f acc[2][4] = {};
  for (int k0 = 0; k0 < C_; k0 += 32) {
    v16bf aF0 = load_a_frag(A, C_, m0,      k0, lane);
    v16bf aF1 = load_a_frag(A, C_, m0 + 16, k0, lane);
#pragma unroll
    for (int f = 0; f < 4; ++f) {
      // B-fragment: lane L = N column (L%16), element i = K row in A-fragment order
      v16bf bF;
#pragma unroll
      for (int i = 0; i < 16; ++i)
        bF[i] = W[(size_t)(k0 + ((lane >> 4) << 3) + (i & 7) + ((i >> 3) << 4)) * C_ + n0 + f * 16 + (lane & 15)];
      acc[0][f] = __builtin_amdgcn_wmma_f32_16x16x32_bf16(false, aF0, false, bF,
                                                          (short)0, acc[0][f], false, false);
      acc[1][f] = __builtin_amdgcn_wmma_f32_16x16x32_bf16(false, aF1, false, bF,
                                                          (short)0, acc[1][f], false, false);
    }
  }

  const int half8 = (lane >> 4) << 3;   // C/D frag: lane = col, vgpr v = row (+8 for hi half)
#pragma unroll
  for (int r = 0; r < 2; ++r) {
#pragma unroll
    for (int f = 0; f < 4; ++f) {
#pragma unroll
      for (int v = 0; v < 8; ++v) {
        int n = n0 + f * 16 + (lane & 15);
        int m = m0 + r * 16 + v + half8;
        float val = acc[r][f][v] + bias[n];
        int b = m / T_, t = m % T_;
        int h = n / D_, d = n % D_;
        if (mode == 0 || mode == 2) {
          dst_bf[((size_t)(b * H_ + h) * T_ + t) * D_ + d] = (__bf16)val;
        } else if (mode == 1) {
          dst_bf[((size_t)(b * H_ + h) * D_ + d) * T_ + t] = (__bf16)val;
        } else {
          dst_f[(size_t)m * C_ + n] = val;
        }
      }
    }
  }
}

// ---------------------------------------------------------------------------
// Flash attention: one wave (blockDim=32) per 16-query tile per (b,h).
// 64 keys per iteration: 8 WMMAs for S = QK^T, one softmax-butterfly pass,
// 8 WMMAs for O += P V.  The next iteration's K/V tiles are prefetched
// (global_prefetch_b8) at the top of each iteration since the LDS bounce +
// barrier prevents the compiler from hoisting those loads itself.
// ---------------------------------------------------------------------------
__global__ void __launch_bounds__(32)
flash_attn_wmma(const __bf16* __restrict__ q,   // (B,H,T,D)
                const __bf16* __restrict__ kt,  // (B,H,D,T)
                const __bf16* __restrict__ v,   // (B,H,T,D)
                __bf16* __restrict__ y)         // (B,T,H,D) == (B,T,C)
{
  __shared__ __bf16 lds_p[16 * 64];     // P tile bounce buffer (single wave / block)
  const int lane = threadIdx.x & 31;
  const int qt0  = blockIdx.x * 16;
  const int bh   = blockIdx.y;
  const int b    = bh / H_;
  const int h    = bh % H_;
  const __bf16* qb = q  + (size_t)bh * T_ * D_;
  const __bf16* kb = kt + (size_t)bh * D_ * T_;
  const __bf16* vb = v  + (size_t)bh * T_ * D_;

  const float scale = 0.125f;           // 1/sqrt(64)
  const int half8 = (lane >> 4) << 3;
  const int col   = lane & 15;

  v16bf qA[2];
#pragma unroll
  for (int c = 0; c < 2; ++c) qA[c] = load_a_frag(qb, D_, qt0, c * 32, lane);

  v8f o[4] = {};
  float m_i[8], l_i[8];
#pragma unroll
  for (int vv = 0; vv < 8; ++vv) { m_i[vv] = -1e30f; l_i[vv] = 0.0f; }

  for (int kt0 = 0; kt0 < qt0 + 16; kt0 += 64) {
    // ---- prefetch next key tile (K rows + V rows) while we work on this one
    if (kt0 + 64 < qt0 + 16) {
      int nk = kt0 + 64;
#pragma unroll
      for (int c = 0; c < 2; ++c)
        __builtin_prefetch(kb + (size_t)(c * 32 + lane) * T_ + nk, 0, 1);
#pragma unroll
      for (int kc = 0; kc < 2; ++kc)
        __builtin_prefetch(vb + (size_t)(nk + kc * 32 + lane) * D_, 0, 1);
    }

    // ---- S = Q @ K^T : four 16-key fragments, K-dim = D in two 32-chunks ----
    v8f s[4];
#pragma unroll
    for (int sfr = 0; sfr < 4; ++sfr) {
      v8f a = {};
#pragma unroll
      for (int c = 0; c < 2; ++c) {
        v16bf bF;
#pragma unroll
        for (int i = 0; i < 16; ++i)
          bF[i] = kb[(size_t)(c * 32 + ((lane >> 4) << 3) + (i & 7) + ((i >> 3) << 4)) * T_ + (kt0 + sfr * 16 + col)];
        a = __builtin_amdgcn_wmma_f32_16x16x32_bf16(false, qA[c], false, bF,
                                                    (short)0, a, false, false);
      }
      s[sfr] = a;
    }
    // ---- scale + causal mask ----
#pragma unroll
    for (int sfr = 0; sfr < 4; ++sfr) {
      int kidx = kt0 + sfr * 16 + col;
#pragma unroll
      for (int vv = 0; vv < 8; ++vv) {
        int qidx = qt0 + vv + half8;
        s[sfr][vv] = (kidx <= qidx) ? s[sfr][vv] * scale : -1e30f;
      }
    }
    // ---- online softmax (one butterfly pass over 64 keys) ----
    float alpha[8];
#pragma unroll
    for (int vv = 0; vv < 8; ++vv) {
      float r = fmaxf(fmaxf(s[0][vv], s[1][vv]), fmaxf(s[2][vv], s[3][vv]));
#pragma unroll
      for (int off = 8; off >= 1; off >>= 1) r = fmaxf(r, __shfl_xor(r, off, 32));
      float m_new = fmaxf(m_i[vv], r);
      alpha[vv] = __expf(m_i[vv] - m_new);
      float rs = 0.0f;
#pragma unroll
      for (int sfr = 0; sfr < 4; ++sfr) {
        float p = __expf(s[sfr][vv] - m_new);
        s[sfr][vv] = p;
        rs += p;
      }
#pragma unroll
      for (int off = 8; off >= 1; off >>= 1) rs += __shfl_xor(rs, off, 32);
      l_i[vv] = l_i[vv] * alpha[vv] + rs;
      m_i[vv] = m_new;
    }
#pragma unroll
    for (int f = 0; f < 4; ++f)
#pragma unroll
      for (int vv = 0; vv < 8; ++vv) o[f][vv] *= alpha[vv];

    // ---- P (C/D layout) -> LDS -> A-fragment layout (16x64) ----
#pragma unroll
    for (int sfr = 0; sfr < 4; ++sfr)
#pragma unroll
      for (int vv = 0; vv < 8; ++vv)
        lds_p[(vv + half8) * 64 + sfr * 16 + col] = (__bf16)s[sfr][vv];
    __syncthreads();
    v16bf pA[2];
#pragma unroll
    for (int kc = 0; kc < 2; ++kc) {
      const __bf16* p = lds_p + (size_t)col * 64 + kc * 32 + ((lane >> 4) << 3);
      v8bf lo = *(const v8bf*)p;
      v8bf hi = *(const v8bf*)(p + 16);
      pA[kc] = __builtin_shufflevector(lo, hi, 0,1,2,3,4,5,6,7,8,9,10,11,12,13,14,15);
    }
#pragma unroll
    for (int f = 0; f < 4; ++f) {
#pragma unroll
      for (int kc = 0; kc < 2; ++kc) {
        // V B-fragment: lane L = d column (L%16), element i = key row in A-fragment order
        v16bf vF;
#pragma unroll
        for (int i = 0; i < 16; ++i)
          vF[i] = vb[(size_t)(kt0 + kc * 32 + ((lane >> 4) << 3) + (i & 7) + ((i >> 3) << 4)) * D_ + f * 16 + col];
        o[f] = __builtin_amdgcn_wmma_f32_16x16x32_bf16(false, pA[kc], false, vF,
                                                       (short)0, o[f], false, false);
      }
    }
    __syncthreads();
  }

  // ---- normalize + store y as (B,T,H,D) ----
#pragma unroll
  for (int vv = 0; vv < 8; ++vv) {
    float inv = 1.0f / l_i[vv];
    int t = qt0 + vv + half8;
#pragma unroll
    for (int f = 0; f < 4; ++f) {
      int d = f * 16 + col;
      y[(((size_t)b * T_ + t) * H_ + h) * D_ + d] = (__bf16)(o[f][vv] * inv);
    }
  }
}

// ---------------------------------------------------------------------------
extern "C" void kernel_launch(void* const* d_in, const int* in_sizes, int n_in,
                              void* d_out, int out_size, void* d_ws, size_t ws_size,
                              hipStream_t stream) {
  (void)in_sizes; (void)n_in; (void)out_size; (void)ws_size;
  const float* x  = (const float*)d_in[0];
  const float* Wq = (const float*)d_in[1];
  const float* bq = (const float*)d_in[2];
  const float* Wk = (const float*)d_in[3];
  const float* bk = (const float*)d_in[4];
  const float* Wv = (const float*)d_in[5];
  const float* bv = (const float*)d_in[6];
  const float* Wp = (const float*)d_in[7];
  const float* bp = (const float*)d_in[8];
  float* out = (float*)d_out;

  const size_t nx = (size_t)B_ * T_ * C_;   // 8 Mi elements
  const size_t nw = (size_t)C_ * C_;        // 1 Mi elements
  __bf16* x_bf  = (__bf16*)d_ws;
  __bf16* wq_bf = x_bf  + nx;
  __bf16* wk_bf = wq_bf + nw;
  __bf16* wv_bf = wk_bf + nw;
  __bf16* wp_bf = wv_bf + nw;
  __bf16* q_bf  = wp_bf + nw;
  __bf16* kt_bf = q_bf  + nx;
  __bf16* v_bf  = kt_bf + nx;
  __bf16* y_bf  = v_bf  + nx;

  cvt_bf16<<<(int)((nx + 255) / 256), 256, 0, stream>>>(x,  x_bf,  (int)nx);
  cvt_bf16<<<(int)((nw + 255) / 256), 256, 0, stream>>>(Wq, wq_bf, (int)nw);
  cvt_bf16<<<(int)((nw + 255) / 256), 256, 0, stream>>>(Wk, wk_bf, (int)nw);
  cvt_bf16<<<(int)((nw + 255) / 256), 256, 0, stream>>>(Wv, wv_bf, (int)nw);
  cvt_bf16<<<(int)((nw + 255) / 256), 256, 0, stream>>>(Wp, wp_bf, (int)nw);

  const int waves  = (B_ * T_ / 32) * (C_ / 64);  // 4096 wave-tiles (32x64 each)
  const int blocks = waves / 8;                    // 8 waves / 256-thread block
  gemm_bf16_wmma<<<blocks, 256, 0, stream>>>(x_bf, wq_bf, bq, q_bf,  nullptr, 0);
  gemm_bf16_wmma<<<blocks, 256, 0, stream>>>(x_bf, wk_bf, bk, kt_bf, nullptr, 1);
  gemm_bf16_wmma<<<blocks, 256, 0, stream>>>(x_bf, wv_bf, bv, v_bf,  nullptr, 2);

  dim3 ag(T_ / 16, B_ * H_);
  flash_attn_wmma<<<ag, 32, 0, stream>>>(q_bf, kt_bf, v_bf, y_bf);

  gemm_bf16_wmma<<<blocks, 256, 0, stream>>>(y_bf, wp_bf, bp, nullptr, out, 3);
}
